// MesoNet_51771535786249
// MI455X (gfx1250) — matrix operand order repaired and checked by reading, not compiled, not run on hardware
//
#include <hip/hip_runtime.h>

// MesoNet forward for MI455X (gfx1250, wave32, WMMA).
// All GEMMs (incl. the edge-conditioned NNConv message einsum, rewritten as
// [h (x) x_src | x_src] @ [W2_reshaped ; eb2]) run on v_wmma_f32_16x16x32_f16.
// B operands are stored fragment-major (transposed) so each lane's 16 halves
// are one aligned v16h load (2x b128) instead of 16 strided 16-bit loads.

typedef __attribute__((ext_vector_type(16))) _Float16 v16h;
typedef __attribute__((ext_vector_type(8)))  float    v8f;

#define kN   6000
#define kE   12000
#define kG   2000
#define kA2G 8000
#define kB   32
#define kDIN 214
#define kHD  160
#define kFD  32

__device__ __forceinline__ float sigf(float x) { return 1.f / (1.f + __expf(-x)); }

__device__ __forceinline__ float act_f(float v, int act) {
  if (act == 1) return fmaxf(v, 0.f);
  if (act == 2) return tanhf(v);
  if (act == 3) return sigf(v);
  return v;
}

// ---------------------------------------------------------------------------
// Generic WMMA GEMM: C[M,Nc] = act( A[M,K] @ B[K,Nc] (+C if beta) (+bias)
//                                   (+ addv/max(cnt,1)) )
// 128 threads = 4 waves. Block tile 32(M) x 64(N); each wave owns a 16-col
// slice and two 16x16 accumulators, so each B fragment feeds two WMMAs.
// A staged [32][32]; B staged transposed [64 cols][32 k] so a lane's B
// fragment is one aligned 32B LDS read.
// ---------------------------------------------------------------------------
__global__ void gemm_wmma_k(const float* __restrict__ A, int lda,
                            const float* __restrict__ B, int ldb,
                            float* __restrict__ C, int ldc,
                            int M, int Nc, int K,
                            const float* __restrict__ bias,
                            const float* __restrict__ addv, int ldadd,
                            const float* __restrict__ cnt,
                            int beta, int act) {
  __shared__ alignas(32) _Float16 As[32][32];
  __shared__ alignas(32) _Float16 Bs[64][32];  // [col][k]
  const int tid  = threadIdx.x;
  const int lane = tid & 31, wave = tid >> 5;
  const int g = lane >> 4, l16 = lane & 15;
  const int m0 = blockIdx.x * 32;
  const int n0 = blockIdx.y * 64;
  const int colbase = wave * 16;
  v8f acc[2] = {{}, {}};
  for (int kb = 0; kb < K; kb += 32) {
    __syncthreads();
    // stage A tile (32 rows x 32 k), coalesced along k
    for (int idx = tid; idx < 32 * 32; idx += 128) {
      int mm = idx >> 5, kk = idx & 31;
      int gm = m0 + mm, gk = kb + kk;
      float v = (gm < M && gk < K) ? A[(size_t)gm * lda + gk] : 0.f;
      As[mm][kk] = (_Float16)v;
    }
    // stage B tile (32 k x 64 cols), coalesced along n, stored transposed
    for (int idx = tid; idx < 32 * 64; idx += 128) {
      int kk = idx >> 6, cc = idx & 63;
      int gk = kb + kk, gc = n0 + cc;
      float v = (gk < K && gc < Nc) ? B[(size_t)gk * ldb + gc] : 0.f;
      Bs[cc][kk] = (_Float16)v;
    }
    // speculative prefetch of next K tile (global_prefetch_b8)
    if (kb + 32 < K) {
      int mm = m0 + (tid >> 2);
      if (mm < M) __builtin_prefetch(&A[(size_t)mm * lda + kb + 32], 0, 1);
      int pk = kb + 32 + (tid >> 6);
      if (pk < K) __builtin_prefetch(&B[(size_t)pk * ldb + n0 + (tid & 63)], 0, 1);
    }
    __syncthreads();
    v16h a0, a1;
#pragma unroll
    for (int j = 0; j < 16; ++j) {
      int ka = g * 16 + j;
      a0[j] = As[l16][ka];
      a1[j] = As[16 + l16][ka];
    }
    // lane's B fragment: K rows g*16..g*16+15 of its column -> contiguous 32B
    v16h b = *(const v16h*)(&Bs[colbase + l16][g * 16]);
    acc[0] = __builtin_amdgcn_wmma_f32_16x16x32_f16(false, a0, false, b,
                                                    (short)0, acc[0], false, false);
    acc[1] = __builtin_amdgcn_wmma_f32_16x16x32_f16(false, a1, false, b,
                                                    (short)0, acc[1], false, false);
  }
  int col = n0 + colbase + l16;
  if (col < Nc) {
#pragma unroll
    for (int t = 0; t < 2; ++t) {
#pragma unroll
      for (int r = 0; r < 8; ++r) {
        int row = m0 + t * 16 + g * 8 + r;
        if (row < M) {
          size_t ci = (size_t)row * ldc + col;
          float v = acc[t][r];
          if (beta) v += C[ci];
          if (bias) v += bias[col];
          if (addv) {
            float av = addv[(size_t)row * ldadd + col];
            if (cnt) av /= fmaxf(cnt[row], 1.f);
            v += av;
          }
          C[ci] = act_f(v, act);
        }
      }
    }
  }
}

// ---------------------------------------------------------------------------
// Pack [W2 reshaped ; eb2] into f16, TRANSPOSED: wbigT[o][k], k zero-padded
// to Kpad. Row k of the logical B matrix:
//   k = f*din+d       -> ew2[f, d*dout+o]
//   k = FD*din+d      -> eb2[d*dout+o]
// ---------------------------------------------------------------------------
__global__ void pack_wbig_k(const float* __restrict__ ew2,
                            const float* __restrict__ eb2,
                            int din, int dout, int Kpad,
                            _Float16* __restrict__ wbigT) {
  long idx = (long)blockIdx.x * blockDim.x + threadIdx.x;
  long total = (long)Kpad * dout;
  if (idx >= total) return;
  int o = (int)(idx / Kpad), k = (int)(idx - (long)o * Kpad);
  int fdx = kFD * din;
  float v = 0.f;
  if (k < fdx) {
    int f = k / din, d = k - f * din;
    v = ew2[(size_t)f * din * dout + (size_t)d * dout + o];
  } else if (k < fdx + din) {
    v = eb2[(size_t)(k - fdx) * dout + o];
  }
  wbigT[idx] = (_Float16)v;
}

// ---------------------------------------------------------------------------
// Fused NNConv message: msg[e,o] = sum_{f,d} h[e,f]*xs[e,d]*W2[f,d,o]
//                                 + sum_d xs[e,d]*eb2[d,o]
// 16 edges / block; DOUT/16 waves. A fragments of [h (x) xs | xs] generated
// on the fly from LDS; B fragments are single aligned v16h loads from the
// transposed packed weights (2x global_load_b128 per WMMA).
// ---------------------------------------------------------------------------
template <int DIN, int DOUT>
__global__ void conv_msg_wmma_k(const float* __restrict__ xin, int xstride,
                                const float* __restrict__ ea,
                                const float* __restrict__ ew1,
                                const float* __restrict__ eb1,
                                const _Float16* __restrict__ wbigT,
                                const int* __restrict__ src,
                                float* __restrict__ msg) {
  constexpr int Ktot = kFD * DIN + DIN;
  constexpr int KPAD = ((Ktot + 31) / 32) * 32;
  __shared__ float hs[16][kFD];
  __shared__ float xs[16][DIN];
  const int tid = threadIdx.x;
  const int e0 = blockIdx.x * 16;
  for (int idx = tid; idx < 16 * DIN; idx += blockDim.x) {
    int le = idx / DIN, d = idx - le * DIN;
    xs[le][d] = xin[(size_t)src[e0 + le] * xstride + d];
  }
  for (int idx = tid; idx < 16 * kFD; idx += blockDim.x) {
    int le = idx >> 5, f = idx & 31;
    const float* er = ea + (size_t)(e0 + le) * 6;
    float s = eb1[f];
#pragma unroll
    for (int i = 0; i < 6; ++i) s += er[i] * ew1[i * kFD + f];
    hs[le][f] = fmaxf(s, 0.f);
  }
  __syncthreads();
  const int lane = tid & 31, wave = tid >> 5;
  const int g = lane >> 4, l16 = lane & 15;
  const int colbase = wave * 16;
  constexpr int fdx = kFD * DIN;
  // this lane's column of the packed B matrix, fragment-major
  const _Float16* bcol = wbigT + (size_t)(colbase + l16) * KPAD + g * 16;
  v8f acc = {};
  for (int kb = 0; kb < Ktot; kb += 32) {
    if (kb + 32 < Ktot) __builtin_prefetch(bcol + kb + 32, 0, 1);
    v16h a;
#pragma unroll
    for (int j = 0; j < 16; ++j) {
      int ka = kb + g * 16 + j;
      float av = 0.f;
      if (ka < fdx) av = hs[l16][ka / DIN] * xs[l16][ka % DIN];
      else if (ka < Ktot) av = xs[l16][ka - fdx];
      a[j] = (_Float16)av;
    }
    v16h b = *(const v16h*)(bcol + kb);  // 32B aligned: KPAD%32==0, kb%32==0
    acc = __builtin_amdgcn_wmma_f32_16x16x32_f16(false, a, false, b,
                                                 (short)0, acc, false, false);
  }
#pragma unroll
  for (int r = 0; r < 8; ++r) {
    int row = g * 8 + r;
    msg[(size_t)(e0 + row) * DOUT + colbase + l16] = acc[r];
  }
}

// ---------------------------------------------------------------------------
// Segment ops + elementwise helpers
// ---------------------------------------------------------------------------
__global__ void seg_accum_k(const float* __restrict__ vals, int ldv,
                            const int* __restrict__ gidx,
                            const int* __restrict__ seg,
                            int R, int D, float* __restrict__ sum,
                            float* __restrict__ cnt) {
  int i = blockIdx.x;
  if (i >= R) return;
  int r = gidx ? gidx[i] : i;
  int s = seg[i];
  const float* vr = vals + (size_t)r * ldv;
  for (int d = threadIdx.x; d < D; d += blockDim.x)
    atomicAdd(&sum[(size_t)s * D + d], vr[d]);
  if (threadIdx.x == 0) atomicAdd(&cnt[s], 1.f);
}

__global__ void seg_div_k(const float* __restrict__ sum,
                          const float* __restrict__ cnt,
                          int S, int D, float* __restrict__ out) {
  long idx = (long)blockIdx.x * blockDim.x + threadIdx.x;
  long tot = (long)S * D;
  if (idx >= tot) return;
  int s = (int)(idx / D);
  out[idx] = sum[idx] / fmaxf(cnt[s], 1.f);
}

__global__ void fill_k(float* p, long n, float v) {
  long i = (long)blockIdx.x * blockDim.x + threadIdx.x;
  long st = (long)gridDim.x * blockDim.x;
  for (; i < n; i += st) p[i] = v;
}

__global__ void copy_k(float* d, const float* s, long n) {
  long i = (long)blockIdx.x * blockDim.x + threadIdx.x;
  long st = (long)gridDim.x * blockDim.x;
  for (; i < n; i += st) d[i] = s[i];
}

__global__ void copy_cols_k(float* dst, int ldd, int coloff,
                            const float* src, int lds, int R, int Cc) {
  long idx = (long)blockIdx.x * blockDim.x + threadIdx.x;
  long tot = (long)R * Cc;
  long st = (long)gridDim.x * blockDim.x;
  for (; idx < tot; idx += st) {
    int r = (int)(idx / Cc), c = (int)(idx - (long)r * Cc);
    dst[(size_t)r * ldd + coloff + c] = src[(size_t)r * lds + c];
  }
}

__global__ void gather_c3_k(const float* __restrict__ x, float* __restrict__ c3, int n) {
  int i = blockIdx.x * blockDim.x + threadIdx.x;
  if (i >= n) return;
  const float* xr = x + (size_t)i * kDIN;
  c3[i * 3 + 0] = xr[131];  // g_feat[:, GD]
  c3[i * 3 + 1] = xr[172];  // G_[:, GD]
  c3[i * 3 + 2] = xr[173];  // G_[:, GD+CD]
}

__global__ void cfc_combine_k(float* __restrict__ h, const float* __restrict__ f1,
                              const float* __restrict__ f2, const float* __restrict__ t,
                              long n) {
  long i = (long)blockIdx.x * blockDim.x + threadIdx.x;
  long st = (long)gridDim.x * blockDim.x;
  for (; i < n; i += st) h[i] = f1[i] * (1.f - t[i]) + t[i] * f2[i];
}

__global__ void muladd_k(float* __restrict__ out, const float* __restrict__ a,
                         const float* __restrict__ x, const float* __restrict__ b,
                         long n) {  // out = a*x + b
  long i = (long)blockIdx.x * blockDim.x + threadIdx.x;
  long st = (long)gridDim.x * blockDim.x;
  for (; i < n; i += st) out[i] = a[i] * x[i] + b[i];
}

__global__ void film2_k(float* __restrict__ out, const float* __restrict__ t1,
                        const float* __restrict__ t2, const float* __restrict__ x,
                        long n) {  // out = t1*x + t2 + x
  long i = (long)blockIdx.x * blockDim.x + threadIdx.x;
  long st = (long)gridDim.x * blockDim.x;
  for (; i < n; i += st) out[i] = t1[i] * x[i] + t2[i] + x[i];
}

// ---- Set2Set pieces ----
__global__ void lstm_k(const float* __restrict__ gates, float* __restrict__ c,
                       float* __restrict__ h) {
  int idx = blockIdx.x * blockDim.x + threadIdx.x;
  if (idx >= kB * kHD) return;
  int r = idx / kHD, d = idx - r * kHD;
  const float* gr = gates + (size_t)r * 4 * kHD;
  float i_ = sigf(gr[d]);
  float f_ = sigf(gr[kHD + d]);
  float g_ = tanhf(gr[2 * kHD + d]);
  float o_ = sigf(gr[3 * kHD + d]);
  float cn = f_ * c[idx] + i_ * g_;
  c[idx] = cn;
  h[idx] = o_ * tanhf(cn);
}

__global__ void edot_k(const float* __restrict__ x, const float* __restrict__ h,
                       const int* __restrict__ batch, float* __restrict__ e, int n) {
  int i = blockIdx.x * blockDim.x + threadIdx.x;
  if (i >= n) return;
  const float* xr = x + (size_t)i * kHD;
  const float* hr = h + (size_t)batch[i] * kHD;
  float s = 0.f;
  for (int d = 0; d < kHD; ++d) s += xr[d] * hr[d];
  e[i] = s;
}

__device__ __forceinline__ unsigned f2o(float f) {
  unsigned u = __float_as_uint(f);
  return (u & 0x80000000u) ? ~u : (u | 0x80000000u);
}
__device__ __forceinline__ float o2f(unsigned u) {
  u = (u & 0x80000000u) ? (u & 0x7FFFFFFFu) : ~u;
  return __uint_as_float(u);
}

__global__ void segmax_k(const float* __restrict__ e, const int* __restrict__ batch,
                         unsigned* __restrict__ mu, int n) {
  int i = blockIdx.x * blockDim.x + threadIdx.x;
  if (i >= n) return;
  atomicMax(&mu[batch[i]], f2o(e[i]));
}

__global__ void softw_k(const float* __restrict__ e, const unsigned* __restrict__ mu,
                        const int* __restrict__ batch, float* __restrict__ wv,
                        float* __restrict__ sbuf, int n) {
  int i = blockIdx.x * blockDim.x + threadIdx.x;
  if (i >= n) return;
  int b = batch[i];
  float w = __expf(e[i] - o2f(mu[b]));
  wv[i] = w;
  atomicAdd(&sbuf[b], w);
}

__global__ void wsum_k(const float* __restrict__ wv, const float* __restrict__ sbuf,
                       const int* __restrict__ batch, const float* __restrict__ xv,
                       float* __restrict__ rbuf, int n) {
  int i = blockIdx.x * blockDim.x + threadIdx.x;
  if (i >= n) return;
  int b = batch[i];
  float a = wv[i] / fmaxf(sbuf[b], 1e-12f);
  const float* xr = xv + (size_t)i * kHD;
  for (int d = 0; d < kHD; ++d) atomicAdd(&rbuf[(size_t)b * kHD + d], a * xr[d]);
}

__global__ void qstar_k(float* __restrict__ q, const float* __restrict__ h,
                        const float* __restrict__ r) {
  int idx = blockIdx.x * blockDim.x + threadIdx.x;
  if (idx >= kB * 2 * kHD) return;
  int row = idx / (2 * kHD), c = idx - row * 2 * kHD;
  q[idx] = (c < kHD) ? h[row * kHD + c] : r[row * kHD + c - kHD];
}

// ---------------------------------------------------------------------------
// Host orchestration
// ---------------------------------------------------------------------------
static const long PSZ[62] = {
    // a11
    192, 32, 41984, 1312, 1312, 32,
    // c1
    192, 32, 819200, 25600, 25600, 160,
    // c2
    192, 32, 819200, 25600, 25600, 160,
    // x2, x22, xm3
    192, 32, 9216, 96, 25600, 160,
    // attn
    2624, 1024, 1024, 1024, 1024, 1024,
    // ncp1
    6633, 6633, 6633, 2144,
    // ncp2
    154560, 154560, 154560, 51200,
    // c1g, c1b, c2g, c2b
    480, 160, 25600, 160, 480, 160, 25600, 160,
    480, 160, 25600, 160, 480, 160, 25600, 160,
    // br, fg, fb
    51200, 160, 25600, 160, 25600, 160,
    // s2s
    307200, 640};

static void launch_gemm(hipStream_t st, const float* A, int lda, const float* B, int ldb,
                        float* C, int ldc, int M, int Nc, int K,
                        const float* bias, const float* addv, int ldadd,
                        const float* cnt, int beta, int act) {
  dim3 grid((M + 31) / 32, (Nc + 63) / 64);
  gemm_wmma_k<<<grid, 128, 0, st>>>(A, lda, B, ldb, C, ldc, M, Nc, K,
                                    bias, addv, ldadd, cnt, beta, act);
}

extern "C" void kernel_launch(void* const* d_in, const int* in_sizes, int n_in,
                              void* d_out, int out_size, void* d_ws, size_t ws_size,
                              hipStream_t stream) {
  const float* x        = (const float*)d_in[0];
  const float* ea       = (const float*)d_in[1];
  const float* x_group  = (const float*)d_in[2];
  const int*   srcp     = (const int*)d_in[n_in - 5];
  const int*   dstp     = (const int*)d_in[n_in - 4];
  const int*   batchp   = (const int*)d_in[n_in - 3];
  const int*   gidxp    = (const int*)d_in[n_in - 2];
  const int*   aidxp    = (const int*)d_in[n_in - 1];

  const float* prm[62];
  if (n_in >= 3 + 62 + 5) {
    for (int i = 0; i < 62; ++i) prm[i] = (const float*)d_in[3 + i];
  } else {
    const float* base = (const float*)d_in[3];
    long off = 0;
    for (int i = 0; i < 62; ++i) { prm[i] = base + off; off += PSZ[i]; }
  }

  // workspace bump allocator
  char* wp = (char*)d_ws;
  auto alloc = [&](size_t bytes) -> void* {
    void* p = wp;
    wp += (bytes + 255) & ~(size_t)255;
    return p;
  };
  _Float16* wbig = (_Float16*)alloc((size_t)5280 * kHD * 2);
  float* msg     = (float*)alloc((size_t)kE * kHD * 4);
  float* segsum  = (float*)alloc((size_t)kN * kHD * 4);
  float* segcnt  = (float*)alloc((size_t)kN * 4);
  float* c3      = (float*)alloc((size_t)kN * 3 * 4);
  float* x1      = (float*)alloc((size_t)kN * 32 * 4);
  float* x2o     = (float*)alloc((size_t)kN * 32 * 4);
  float* tA      = (float*)alloc((size_t)kN * 32 * 4);
  float* inter   = (float*)alloc((size_t)kN * 32 * 4);
  float* glob    = (float*)alloc((size_t)kN * 32 * 4);
  float* grp     = (float*)alloc((size_t)kN * 32 * 4);
  float* hidden  = (float*)alloc((size_t)kN * 67 * 4);
  float* ffa     = (float*)alloc((size_t)kN * 320 * 4);
  float* ffb     = (float*)alloc((size_t)kN * 320 * 4);
  float* ffc     = (float*)alloc((size_t)kN * 320 * 4);
  float* outs    = (float*)alloc((size_t)kN * 96 * 4);
  float* buf96   = (float*)alloc((size_t)kN * 96 * 4);
  float* xm      = (float*)alloc((size_t)kN * kHD * 4);
  float* filma   = (float*)alloc((size_t)kN * kHD * 4);
  float* filmb   = (float*)alloc((size_t)kN * kHD * 4);
  float* xmf     = (float*)alloc((size_t)kN * kHD * 4);
  float* f160    = (float*)alloc((size_t)kN * kHD * 4);
  float* gsum    = (float*)alloc((size_t)kG * kHD * 4);
  float* gcnt    = (float*)alloc((size_t)kG * 4);
  float* gagg    = (float*)alloc((size_t)kG * kHD * 4);
  float* asum    = (float*)alloc((size_t)kN * kHD * 4);
  float* acnt    = (float*)alloc((size_t)kN * 4);
  float* afeat   = (float*)alloc((size_t)kN * kHD * 4);
  float* t1      = (float*)alloc((size_t)kN * kHD * 4);
  float* t2      = (float*)alloc((size_t)kN * kHD * 4);
  float* hidden2 = (float*)alloc((size_t)kN * 320 * 4);
  float* sx      = (float*)alloc((size_t)kN * kHD * 4);
  float* sxin    = (float*)alloc((size_t)kN * kHD * 4);
  float* g2buf   = (float*)alloc((size_t)kN * kHD * 4);
  float* b2buf   = (float*)alloc((size_t)kN * kHD * 4);
  float* sx3     = (float*)alloc((size_t)kN * kHD * 4);
  float* qstar   = (float*)alloc((size_t)kB * 320 * 4);
  float* hst     = (float*)alloc((size_t)kB * kHD * 4);
  float* cst     = (float*)alloc((size_t)kB * kHD * 4);
  float* gates   = (float*)alloc((size_t)kB * 640 * 4);
  float* ebuf    = (float*)alloc((size_t)kN * 4);
  float* wvbuf   = (float*)alloc((size_t)kN * 4);
  float* mubuf   = (float*)alloc((size_t)kB * 4);
  float* sbuf    = (float*)alloc((size_t)kB * 4);
  float* rbuf    = (float*)alloc((size_t)kB * kHD * 4);
  float* bsum    = (float*)alloc((size_t)kB * 32 * 4);
  float* bcnt    = (float*)alloc((size_t)kB * 4);
  (void)ws_size; (void)in_sizes; (void)out_size;

  auto fillz = [&](float* p, long n) {
    int blocks = (int)((n + 255) / 256); if (blocks > 4096) blocks = 4096;
    fill_k<<<blocks, 256, 0, stream>>>(p, n, 0.f);
  };
  auto ew = [&](long n) {
    int blocks = (int)((n + 255) / 256); if (blocks > 4096) blocks = 4096;
    return blocks;
  };
  auto gemm = [&](const float* A, int lda, const float* B, int ldb, float* C, int ldc,
                  int M, int Nc, int K, const float* bias = nullptr,
                  const float* addv = nullptr, int ldadd = 0,
                  const float* cnt = nullptr, int beta = 0, int act = 0) {
    launch_gemm(stream, A, lda, B, ldb, C, ldc, M, Nc, K, bias, addv, ldadd, cnt, beta, act);
  };

  // NNConv: out = relu( xin@root + seg_mean(msg, dst) + bias )
  auto nnconv = [&](const float* xin, int xstride, int din, int dout,
                    const float* ew1, const float* eb1, const float* ew2,
                    const float* eb2, const float* root, const float* bias_,
                    float* out) {
    int Ktot = kFD * din + din;
    int Kpad = ((Ktot + 31) / 32) * 32;
    long tot = (long)Kpad * dout;
    pack_wbig_k<<<(int)((tot + 255) / 256), 256, 0, stream>>>(ew2, eb2, din, dout, Kpad, wbig);
    if (din == 41) {
      conv_msg_wmma_k<41, 32><<<kE / 16, 64, 0, stream>>>(
          xin, xstride, ea, ew1, eb1, wbig, srcp, msg);
    } else {
      conv_msg_wmma_k<160, 160><<<kE / 16, 320, 0, stream>>>(
          xin, xstride, ea, ew1, eb1, wbig, srcp, msg);
    }
    fillz(segsum, (long)kN * dout);
    fillz(segcnt, kN);
    seg_accum_k<<<kE, 64, 0, stream>>>(msg, dout, nullptr, dstp, kE, dout, segsum, segcnt);
    gemm(xin, xstride, root, dout, out, dout, kN, dout, din, bias_, segsum, dout,
         segcnt, 0, /*relu*/ 1);
  };

  auto film = [&](const float* w1, const float* b1, const float* w2, const float* b2,
                  float* out) {
    gemm(c3, 3, w1, kHD, f160, kHD, kN, kHD, 3, b1, nullptr, 0, nullptr, 0, 1);
    gemm(f160, kHD, w2, kHD, out, kHD, kN, kHD, kHD, b2, nullptr, 0, nullptr, 0, 0);
  };

  // CfC with ncp2 params: updates hidden2 in place from input `in` (N x 160)
  auto cfc2 = [&](const float* in) {
    const float* Ws[3] = {prm[34], prm[35], prm[36]};
    float* ffs[3] = {ffa, ffb, ffc};
    int acts[3] = {2, 2, 3};
    for (int k = 0; k < 3; ++k) {
      gemm(in, kHD, Ws[k], 320, ffs[k], 320, kN, 320, kHD, nullptr, nullptr, 0, nullptr, 0, 0);
      gemm(c3, 3, Ws[k] + 160 * 320, 320, ffs[k], 320, kN, 320, 3, nullptr, nullptr, 0, nullptr, 1, 0);
      gemm(hidden2, 320, Ws[k] + 163 * 320, 320, ffs[k], 320, kN, 320, 320, nullptr, nullptr, 0, nullptr, 1, acts[k]);
    }
    cfc_combine_k<<<ew((long)kN * 320), 256, 0, stream>>>(hidden2, ffa, ffb, ffc, (long)kN * 320);
  };

  // ---- forward pass ----
  gather_c3_k<<<(kN + 255) / 256, 256, 0, stream>>>(x, c3, kN);

  // x1 = relu(nnconv a11)
  nnconv(x, kDIN, 41, 32, prm[0], prm[1], prm[2], prm[3], prm[4], prm[5], x1);

  // x2o = relu(x2 @ x2_w + b)
  gemm(x + 41, kDIN, prm[18], 32, x2o, 32, kN, 32, 6, prm[19], nullptr, 0, nullptr, 0, 1);

  // inter / glob / grp (xattn chains)
  gemm(x + 132, kDIN, prm[24], 32, tA, 32, kN, 32, 82);
  gemm(tA, 32, prm[25], 32, inter, 32, kN, 32, 32);
  gemm(inter, 32, prm[26], 32, tA, 32, kN, 32, 32);
  gemm(tA, 32, prm[27], 32, glob, 32, kN, 32, 32);
  gemm(glob, 32, prm[28], 32, tA, 32, kN, 32, 32);
  gemm(tA, 32, prm[29], 32, grp, 32, kN, 32, 32);

  // hidden = [grp | glob | C]
  copy_cols_k<<<ew((long)kN * 32), 256, 0, stream>>>(hidden, 67, 0, grp, 32, kN, 32);
  copy_cols_k<<<ew((long)kN * 32), 256, 0, stream>>>(hidden, 67, 32, glob, 32, kN, 32);
  copy_cols_k<<<ew((long)kN * 3), 256, 0, stream>>>(hidden, 67, 64, c3, 3, kN, 3);

  // CfC ncp1 x3
  {
    const float* Ws[3] = {prm[30], prm[31], prm[32]};
    int acts[3] = {2, 2, 3};
    float* ffs[3] = {ffa, ffb, ffc};
    const float* xi = x2o;
    int ldxi = 32;
    for (int it = 0; it < 3; ++it) {
      for (int k = 0; k < 3; ++k) {
        gemm(xi, ldxi, Ws[k], 67, ffs[k], 67, kN, 67, 32, nullptr, nullptr, 0, nullptr, 0, 0);
        gemm(hidden, 67, Ws[k] + 32 * 67, 67, ffs[k], 67, kN, 67, 67, nullptr, nullptr, 0, nullptr, 1, acts[k]);
      }
      cfc_combine_k<<<ew((long)kN * 67), 256, 0, stream>>>(hidden, ffa, ffb, ffc, (long)kN * 67);
      gemm(hidden, 67, prm[33], 32, outs + it * 32, 96, kN, 32, 67);
      xi = outs + it * 32;
      ldxi = 96;
    }
  }

  // x2o' = relu(outs @ x22_w + b)
  gemm(outs, 96, prm[20], 96, buf96, 96, kN, 96, 96, prm[21], nullptr, 0, nullptr, 0, 1);

  // xm = relu([x2o' | x1 | glob] @ xm3_w + b)  (3-pass accumulate)
  gemm(buf96, 96, prm[22], kHD, xm, kHD, kN, kHD, 96);
  gemm(x1, 32, prm[22] + 96 * kHD, kHD, xm, kHD, kN, kHD, 32, nullptr, nullptr, 0, nullptr, 1, 0);
  gemm(glob, 32, prm[22] + 128 * kHD, kHD, xm, kHD, kN, kHD, 32, prm[23], nullptr, 0, nullptr, 1, 1);

  // xm_film = film(C,c1g)*xm + film(C,c1b)
  film(prm[38], prm[39], prm[40], prm[41], filma);
  film(prm[42], prm[43], prm[44], prm[45], filmb);
  muladd_k<<<ew((long)kN * kHD), 256, 0, stream>>>(xmf, filma, xm, filmb, (long)kN * kHD);

  // group aggregation + xg_after (written directly into d_out region)
  float* xg_after = (float*)d_out + kB * 2 * kHD;  // offset 10240
  fillz(gsum, (long)kG * kHD);
  fillz(gcnt, kG);
  seg_accum_k<<<kA2G, 64, 0, stream>>>(xmf, kHD, aidxp, gidxp, kA2G, kHD, gsum, gcnt);
  seg_div_k<<<(kG * kHD + 255) / 256, 256, 0, stream>>>(gsum, gcnt, kG, kHD, gagg);
  gemm(x_group, kHD, prm[54], kHD, xg_after, kHD, kG, kHD, kHD);
  gemm(gagg, kHD, prm[54] + 160 * kHD, kHD, xg_after, kHD, kG, kHD, kHD, prm[55], nullptr, 0, nullptr, 1, 1);

  // a_feat = seg_mean(xg_after[g_idx], a_idx)
  fillz(asum, (long)kN * kHD);
  fillz(acnt, kN);
  seg_accum_k<<<kA2G, 64, 0, stream>>>(xg_after, kHD, gidxp, aidxp, kA2G, kHD, asum, acnt);
  seg_div_k<<<(kN * kHD + 255) / 256, 256, 0, stream>>>(asum, acnt, kN, kHD, afeat);

  // xm_film = (afeat@fg+fgb)*xm_film + (afeat@fb+fbb) + xm_film
  gemm(afeat, kHD, prm[56], kHD, t1, kHD, kN, kHD, kHD, prm[57]);
  gemm(afeat, kHD, prm[58], kHD, t2, kHD, kN, kHD, kHD, prm[59]);
  film2_k<<<ew((long)kN * kHD), 256, 0, stream>>>(xmf, t1, t2, xmf, (long)kN * kHD);

  // hidden2 = [xm_film | xm_film]
  copy_cols_k<<<ew((long)kN * kHD), 256, 0, stream>>>(hidden2, 320, 0, xmf, kHD, kN, kHD);
  copy_cols_k<<<ew((long)kN * kHD), 256, 0, stream>>>(hidden2, 320, kHD, xmf, kHD, kN, kHD);

  cfc2(xmf);

  // sx = relu(nnconv c1(xm_film))
  nnconv(xmf, kHD, kHD, kHD, prm[6], prm[7], prm[8], prm[9], prm[10], prm[11], sx);
  cfc2(sx);

  // gamma2/beta2
  film(prm[46], prm[47], prm[48], prm[49], g2buf);
  film(prm[50], prm[51], prm[52], prm[53], b2buf);

  muladd_k<<<ew((long)kN * kHD), 256, 0, stream>>>(sxin, g2buf, sx, b2buf, (long)kN * kHD);
  nnconv(sxin, kHD, kHD, kHD, prm[12], prm[13], prm[14], prm[15], prm[16], prm[17], sx);
  cfc2(sx);

  muladd_k<<<ew((long)kN * kHD), 256, 0, stream>>>(sxin, g2buf, sx, b2buf, (long)kN * kHD);
  nnconv(sxin, kHD, kHD, kHD, prm[12], prm[13], prm[14], prm[15], prm[16], prm[17], sx);

  // sx3 = CfC(ncp2) output projection of updated hidden
  cfc2(sx);
  gemm(hidden2, 320, prm[37], kHD, sx3, kHD, kN, kHD, 320);

  // Set2Set (2 steps)
  fillz(qstar, (long)kB * 320);
  fillz(hst, (long)kB * kHD);
  fillz(cst, (long)kB * kHD);
  for (int step = 0; step < 2; ++step) {
    gemm(qstar, 320, prm[60], 640, gates, 640, kB, 640, 320);
    gemm(hst, kHD, prm[60] + 320 * 640, 640, gates, 640, kB, 640, kHD, prm[61], nullptr, 0, nullptr, 1, 0);
    lstm_k<<<(kB * kHD + 255) / 256, 256, 0, stream>>>(gates, cst, hst);
    edot_k<<<(kN + 255) / 256, 256, 0, stream>>>(sx3, hst, batchp, ebuf, kN);
    fillz(mubuf, kB);
    segmax_k<<<(kN + 255) / 256, 256, 0, stream>>>(ebuf, batchp, (unsigned*)mubuf, kN);
    fillz(sbuf, kB);
    softw_k<<<(kN + 255) / 256, 256, 0, stream>>>(ebuf, (unsigned*)mubuf, batchp, wvbuf, sbuf, kN);
    fillz(rbuf, (long)kB * kHD);
    wsum_k<<<(kN + 255) / 256, 256, 0, stream>>>(wvbuf, sbuf, batchp, sx3, rbuf, kN);
    qstar_k<<<(kB * 320 + 255) / 256, 256, 0, stream>>>(qstar, hst, rbuf);
  }
  copy_k<<<ew((long)kB * 320), 256, 0, stream>>>((float*)d_out, qstar, (long)kB * 320);

  // group = seg_mean(inter, batch)
  fillz(bsum, (long)kB * 32);
  fillz(bcnt, kB);
  seg_accum_k<<<kN, 64, 0, stream>>>(inter, 32, nullptr, batchp, kN, 32, bsum, bcnt);
  seg_div_k<<<(kB * 32 + 255) / 256, 256, 0, stream>>>(
      bsum, bcnt, kB, 32, (float*)d_out + kB * 2 * kHD + (long)kG * kHD);
}
